// ODEFunc_80582176407958
// MI455X (gfx1250) — compile-verified, matrix operand order checked by reading, not run on hardware
//
#include <hip/hip_runtime.h>
#include <hip/hip_bf16.h>
#include <stdint.h>

// ---------------------------------------------------------------------------
// Fused time-conditioned MLP + exact Jacobian-trace (CNF divergence) for
// MI455X (gfx1250, wave32, WMMA, async-to-LDS DMA).
//
//   a1 = z·W1[:D,:] + (b1 + t·W1[D,:])    h1 = silu(a1)  s1 = silu'(a1)
//   a2 = h1·W2 + b2                        h2 = silu(a2)  s2 = silu'(a2)
//   dz = h2·W3 + b3
//   div = s1ᵀ·G·s2,  G[i,k] = W2[i,k]·M[k,i],  M[k,i] = Σⱼ W3[k,j]·W1[j,i]
//   dlogp = -div
// ---------------------------------------------------------------------------

typedef __attribute__((ext_vector_type(16))) __bf16 bf16x16;
typedef __attribute__((ext_vector_type(8)))  __bf16 bf16x8;
typedef __attribute__((ext_vector_type(8)))  float  f32x8;

#define DD      64
#define HH      256
#define ROWS    64     // batch rows per block
#define THREADS 128    // 4 wave32 waves, 16-row strip each

#define WMMA_BF16(A, B, C) \
  __builtin_amdgcn_wmma_f32_16x16x32_bf16(false, (A), false, (B), (short)0, (C), false, false)

// sigmoid with fast v_rcp_f32 (no IEEE div_scale/div_fmas sequence)
__device__ __forceinline__ float fast_sigmoid(float x) {
  return __builtin_amdgcn_rcpf(1.f + __expf(-x));
}

// dword-wise bf16 pair unpack/pack (exact: bf16 = top 16 bits of f32)
__device__ __forceinline__ float bf16lo(uint32_t w) { return __uint_as_float(w << 16); }
__device__ __forceinline__ float bf16hi(uint32_t w) { return __uint_as_float(w & 0xffff0000u); }
__device__ __forceinline__ uint32_t pack2_bf16(float lo, float hi) {
  union { uint32_t u; __bf16 e[2]; } r;
  r.e[0] = (__bf16)lo; r.e[1] = (__bf16)hi;
  return r.u;   // -> v_cvt_pk_bf16_f32
}

// One 16x32 bf16 A/B fragment for v_wmma_f32_16x16x32_bf16 from a row-major
// [row][K] bf16 array: lane L holds row (L&15); elems 0..7 = K[off..off+8),
// elems 8..15 = K[off+16..off+24), off = (L>=16)*8.
__device__ __forceinline__ bf16x16 ldsFrag(const __bf16* p) {
  union { bf16x16 v; bf16x8 h[2]; } u;
  u.h[0] = *(const bf16x8*)(p);
  u.h[1] = *(const bf16x8*)(p + 16);
  return u.v;
}

// Same fragment from global f32 (z) with on-the-fly bf16 conversion.
__device__ __forceinline__ bf16x16 gldZFrag(const float* __restrict__ z,
                                            int m, int kstep, int half) {
  const float* p = z + m * DD + kstep * 32 + half * 8;
  union { bf16x16 v; uint32_t w[8]; } u;
#pragma unroll
  for (int i = 0; i < 4; ++i) u.w[i]     = pack2_bf16(p[2 * i],      p[2 * i + 1]);
#pragma unroll
  for (int i = 0; i < 4; ++i) u.w[4 + i] = pack2_bf16(p[16 + 2 * i], p[16 + 2 * i + 1]);
  return u.v;
}

// CDNA5 async DMA: global -> LDS, 16B per lane, tracked by ASYNCcnt.
__device__ __forceinline__ void async_b128(uint32_t lds_off, uint32_t goff,
                                           const void* gbase) {
  asm volatile("global_load_async_to_lds_b128 %0, %1, %2 offset:0"
               :: "v"(lds_off), "v"(goff),
                  "s"((unsigned long long)(uintptr_t)gbase)
               : "memory");
}
__device__ __forceinline__ void wait_async0() {
  asm volatile("s_wait_asynccnt 0x0" ::: "memory");
}
__device__ __forceinline__ uint32_t lds_off32(const void* p) {
  return (uint32_t)(uintptr_t)p;  // low 32 bits of generic ptr = LDS offset
}

// ---------------------------------------------------------------------------
// Prep: beff1 = b1 + t*W1[D,:]; bf16 transposed weights W1t[H][D], W2t[H][H],
// W3t[D][H]; Gt[k2][i] = W2[i][k2] * Σⱼ W3[k2][j]·W1[j][i].
// ---------------------------------------------------------------------------
__global__ void ode_prep(const float* __restrict__ t,  const float* __restrict__ W1,
                         const float* __restrict__ b1, const float* __restrict__ W2,
                         const float* __restrict__ W3,
                         __bf16* __restrict__ W1t, __bf16* __restrict__ W2t,
                         __bf16* __restrict__ W3t, __bf16* __restrict__ Gt,
                         float* __restrict__ beff1) {
  const int k2 = blockIdx.x;   // 0..255
  const int i  = threadIdx.x;  // 0..255
  float m = 0.f;
#pragma unroll 8
  for (int j = 0; j < DD; ++j) m += W3[k2 * DD + j] * W1[j * HH + i];
  Gt[k2 * HH + i]  = (__bf16)(W2[i * HH + k2] * m);
  W2t[k2 * HH + i] = (__bf16)(W2[i * HH + k2]);
  if (i < DD)  W1t[k2 * DD + i] = (__bf16)(W1[i * HH + k2]);
  if (k2 < DD) W3t[k2 * HH + i] = (__bf16)(W3[i * DD + k2]);
  if (k2 == 0) beff1[i] = b1[i] + t[0] * W1[DD * HH + i];
}

// ---------------------------------------------------------------------------
// Main fused kernel. LDS (320KB = WGP max):
//   sW2 128KB (W2t), sG 128KB (Gt)  -- resident together => layer2 & Q fused
//   sWx  32KB (W1t, later W3t via overlapped async DMA)
//   sS   32KB (a1 pre-activations, later h2)
// ---------------------------------------------------------------------------
__global__ __launch_bounds__(THREADS, 1)
void ode_fused(const float* __restrict__ z,
               const float* __restrict__ b2,  const float* __restrict__ b3,
               const float* __restrict__ beff1,
               const __bf16* __restrict__ W1t, const __bf16* __restrict__ W2t,
               const __bf16* __restrict__ W3t, const __bf16* __restrict__ Gt,
               float* __restrict__ dz, float* __restrict__ dlogp) {
  __shared__ __bf16 sW2[HH * HH];
  __shared__ __bf16 sG [HH * HH];
  __shared__ __bf16 sWx[DD * HH];
  __shared__ __bf16 sS [ROWS * HH];

  const int tid   = threadIdx.x;
  const int wave  = tid >> 5;
  const int lane  = tid & 31;
  const int half  = (lane >> 4) & 1;
  const int nlane = lane & 15;
  const int mrow0 = wave * 16;
  const int gm0   = blockIdx.x * ROWS;

  const uint32_t oW2 = lds_off32(sW2);
  const uint32_t oG  = lds_off32(sG);
  const uint32_t oWx = lds_off32(sWx);

  // ---- DMA W1t (32KB) into sWx; wait; barrier ----
#pragma unroll
  for (int c = 0; c < (DD * HH * 2) / (16 * THREADS); ++c) {
    const uint32_t off = (c * THREADS + tid) * 16;
    async_b128(oWx + off, off, W1t);
  }
  wait_async0();
  __syncthreads();

  // ---- layer 1: a1 = Z·W1t + beff1 (K=64), two tiles per pass so the four
  //      WMMAs form two independent chains (no dependent-WMMA NOP slots).
  //      W2t/Gt DMA issue interleaved branch-free: 8 chunks of each per pass.
  bf16x16 za0 = gldZFrag(z, gm0 + mrow0 + nlane, 0, half);
  bf16x16 za1 = gldZFrag(z, gm0 + mrow0 + nlane, 1, half);

  for (int nt = 0; nt < HH / 16; nt += 2) {
#pragma unroll
    for (int j = 0; j < 8; ++j) {
      const uint32_t off = ((nt * 4 + j) * THREADS + tid) * 16;  // 0..63 chunks
      async_b128(oW2 + off, off, W2t);
      async_b128(oG + off, off, Gt);
    }
    const int n0 = nt * 16 + nlane;
    const int n1 = n0 + 16;
    f32x8 accA = {}, accB = {};
    bf16x16 b00 = ldsFrag(&sWx[n0 * DD + 0 * 32 + half * 8]);
    bf16x16 b10 = ldsFrag(&sWx[n1 * DD + 0 * 32 + half * 8]);
    accA = WMMA_BF16(za0, b00, accA);
    accB = WMMA_BF16(za0, b10, accB);
    bf16x16 b01 = ldsFrag(&sWx[n0 * DD + 1 * 32 + half * 8]);
    bf16x16 b11 = ldsFrag(&sWx[n1 * DD + 1 * 32 + half * 8]);
    accA = WMMA_BF16(za1, b01, accA);
    accB = WMMA_BF16(za1, b11, accB);
    const float biasA = beff1[n0], biasB = beff1[n1];
#pragma unroll
    for (int r = 0; r < 8; ++r) {
      const int row = (mrow0 + r + half * 8) * HH;
      sS[row + n0] = (__bf16)(accA[r] + biasA);
      sS[row + n1] = (__bf16)(accB[r] + biasB);
    }
  }
  wait_async0();      // W2t/Gt fully in LDS
  __syncthreads();    // everyone done with sWx (W1t); a1 visible

  // ---- kick W3t DMA into sWx; drains during the layer-2/Q compute ----
#pragma unroll
  for (int c = 0; c < (DD * HH * 2) / (16 * THREADS); ++c) {
    const uint32_t off = (c * THREADS + tid) * 16;
    async_b128(oWx + off, off, W3t);
  }

  // ---- rebuild h1 / s1 A-fragments in registers (dword-wise bf16 math) ----
  bf16x16 h1f[8], s1f[8];
#pragma unroll
  for (int ks = 0; ks < 8; ++ks) {
    union { bf16x16 v; uint32_t w[8]; } ux, uh, us;
    ux.v = ldsFrag(&sS[(mrow0 + nlane) * HH + ks * 32 + half * 8]);
#pragma unroll
    for (int d = 0; d < 8; ++d) {
      const uint32_t w = ux.w[d];
      const float x0 = bf16lo(w),        x1 = bf16hi(w);
      const float g0 = fast_sigmoid(x0), g1 = fast_sigmoid(x1);
      const float h0 = x0 * g0,          h1 = x1 * g1;
      uh.w[d] = pack2_bf16(h0, h1);
      us.w[d] = pack2_bf16(g0 + h0 * (1.f - g0), g1 + h1 * (1.f - g1));
    }
    h1f[ks] = uh.v;
    s1f[ks] = us.v;
  }

  // ---- fused layer 2 + divergence GEMM, interleaved accumulator chains:
  //      a2 = H1·W2t + b2 ;  Q = S1·Gt ;  dacc += Q ⊙ silu'(a2) ----
  float dacc[8];
#pragma unroll
  for (int r = 0; r < 8; ++r) dacc[r] = 0.f;

  for (int nt = 0; nt < HH / 16; ++nt) {
    const int n = nt * 16 + nlane;
    f32x8 acc2 = {}, accq = {};
#pragma unroll
    for (int ks = 0; ks < 8; ++ks) {
      bf16x16 bw = ldsFrag(&sW2[n * HH + ks * 32 + half * 8]);
      bf16x16 bg = ldsFrag(&sG [n * HH + ks * 32 + half * 8]);
      acc2 = WMMA_BF16(h1f[ks], bw, acc2);   // independent chains alternate:
      accq = WMMA_BF16(s1f[ks], bg, accq);   // fills WMMA->WMMA hazard slots
    }
    const float bias = b2[n];
#pragma unroll
    for (int r = 0; r < 8; ++r) {
      const float x  = acc2[r] + bias;
      const float sg = fast_sigmoid(x);
      const float h2 = x * sg;
      sS[(mrow0 + r + half * 8) * HH + n] = (__bf16)h2;   // h2 (reuse sS)
      dacc[r] += accq[r] * (sg + h2 * (1.f - sg));         // q * silu'
    }
  }

  // ---- divergence row-sum: butterfly over the 16 lanes of each half ----
#pragma unroll
  for (int r = 0; r < 8; ++r) {
    float v = dacc[r];
    v += __shfl_xor(v, 1);
    v += __shfl_xor(v, 2);
    v += __shfl_xor(v, 4);
    v += __shfl_xor(v, 8);
    dacc[r] = v;
  }
  float dval = dacc[0];
#pragma unroll
  for (int r = 1; r < 8; ++r) dval = (nlane == r) ? dacc[r] : dval;
  if (nlane < 8) dlogp[gm0 + mrow0 + half * 8 + nlane] = -dval;

  wait_async0();      // W3t in LDS
  __syncthreads();    // all h2 tiles visible, sWx reads safe

  // ---- layer 3: dz = H2·W3t + b3 (N=64, K=256), 2 tiles per pass ----
  bf16x16 h2f[8];
#pragma unroll
  for (int ks = 0; ks < 8; ++ks)
    h2f[ks] = ldsFrag(&sS[(mrow0 + nlane) * HH + ks * 32 + half * 8]);

  for (int np = 0; np < 2; ++np) {
    const int n0 = (np * 2 + 0) * 16 + nlane;
    const int n1 = (np * 2 + 1) * 16 + nlane;
    f32x8 accA = {}, accB = {};
#pragma unroll
    for (int ks = 0; ks < 8; ++ks) {
      bf16x16 b0 = ldsFrag(&sWx[n0 * HH + ks * 32 + half * 8]);
      bf16x16 b1 = ldsFrag(&sWx[n1 * HH + ks * 32 + half * 8]);
      accA = WMMA_BF16(h2f[ks], b0, accA);
      accB = WMMA_BF16(h2f[ks], b1, accB);
    }
    const float biasA = b3[n0], biasB = b3[n1];
#pragma unroll
    for (int r = 0; r < 8; ++r) {
      const int m = gm0 + mrow0 + r + half * 8;
      dz[m * DD + n0] = accA[r] + biasA;
      dz[m * DD + n1] = accB[r] + biasB;
    }
  }
}

// ---------------------------------------------------------------------------
extern "C" void kernel_launch(void* const* d_in, const int* in_sizes, int n_in,
                              void* d_out, int out_size, void* d_ws, size_t ws_size,
                              hipStream_t stream) {
  const float* t  = (const float*)d_in[0];  // [1]
  const float* z  = (const float*)d_in[1];  // [B,64]
  const float* W1 = (const float*)d_in[3];  // [65,256]
  const float* b1 = (const float*)d_in[4];  // [256]
  const float* W2 = (const float*)d_in[5];  // [256,256]
  const float* b2 = (const float*)d_in[6];  // [256]
  const float* W3 = (const float*)d_in[7];  // [256,64]
  const float* b3 = (const float*)d_in[8];  // [64]

  const int B = in_sizes[1] / DD;

  char* ws = (char*)d_ws;
  __bf16* W1t   = (__bf16*)(ws + 0);        //  32768 B
  __bf16* W2t   = (__bf16*)(ws + 32768);    // 131072 B
  __bf16* W3t   = (__bf16*)(ws + 163840);   //  32768 B
  __bf16* Gt    = (__bf16*)(ws + 196608);   // 131072 B
  float*  beff1 = (float*)(ws + 327680);    //   1024 B

  float* dz    = (float*)d_out;        // [B,64]
  float* dlogp = dz + (size_t)B * DD;  // [B]

  ode_prep<<<HH, HH, 0, stream>>>(t, W1, b1, W2, W3, W1t, W2t, W3t, Gt, beff1);
  ode_fused<<<B / ROWS, THREADS, 0, stream>>>(z, b2, b3, beff1,
                                              W1t, W2t, W3t, Gt, dz, dlogp);
}
